// LocalMP_14817637171211
// MI455X (gfx1250) — matrix operand order repaired and checked by reading, not compiled, not run on hardware
//
#include <hip/hip_runtime.h>

// ---------------- CDNA5 WMMA types ----------------
typedef __attribute__((ext_vector_type(16))) _Float16 v16h;
typedef __attribute__((ext_vector_type(8)))  _Float16 v8h;
typedef __attribute__((ext_vector_type(8)))  float    v8f;

static constexpr int Nn = 100000;
static constexpr int Ee = 1200000;
static constexpr int Hh = 64;
static constexpr int NCc = 10000;
static constexpr int Gg = 2048;
static constexpr int Ll = 3;

// ========== weight pack: f32 W[K,Nout] -> f16 WMMA B-fragment order ==========
// pk[((kc*NT + nt)*32 + lane)*16 + j] = W[kc*32 + (lane<16?0:16) + j][nt*16 + (lane&15)]
// so each lane's 16-half fragment is one contiguous 32-byte load in the GEMM.
__global__ void lmp_pack_w16(const float* __restrict__ W, _Float16* __restrict__ pk,
                             int K, int Nout) {
    int i = blockIdx.x * blockDim.x + threadIdx.x;
    if (i >= K * Nout) return;
    int j    = i & 15;
    int lane = (i >> 4) & 31;
    int rest = i >> 9;
    int NT   = Nout >> 4;
    int nt   = rest % NT;
    int kc   = rest / NT;
    int k    = (kc << 5) + ((lane < 16) ? 0 : 16) + j;
    int col  = nt * 16 + (lane & 15);
    pk[i] = (_Float16)W[(size_t)k * Nout + col];
}

// ========== WMMA GEMM: C[M,Nout] = A[M,K](f16) @ Bpk(f16, packed) + bias ==========
// K is compile-time (64 or 128) -> branch-free fully-unrolled WMMA inner loop.
// One wave per (2 row tiles x 1 col tile). B fragments for full K held in regs.
// M a multiple of 16 (100000, 10000 both are).
template <int K>
__global__ void lmp_gemm_wmma(const _Float16* __restrict__ A,
                              const _Float16* __restrict__ Bpk,
                              const float* __restrict__ bias,
                              float* __restrict__ C,
                              int M, int Nout) {
    constexpr int KC = K >> 5;                 // 2 or 4 K-chunks
    const int lane = threadIdx.x;              // 0..31, wave32
    const int Mt = M >> 4;
    const int NT = Nout >> 4;
    const int nt = blockIdx.y;
    const int l15 = lane & 15;
    const int kbaseA = (lane < 16) ? 0 : 16;   // contiguous A-frag layout (matches packed B)

    // B fragments for this column tile: contiguous 32B per lane per K-chunk
    v16h bf[KC];
    #pragma unroll
    for (int kc = 0; kc < KC; ++kc)
        bf[kc] = *(const v16h*)(Bpk + ((size_t)(kc * NT + nt) << 9) + (lane << 4));

    const float bv = bias[nt * 16 + l15];

    #pragma unroll
    for (int r = 0; r < 2; ++r) {
        int mt = blockIdx.x * 2 + r;
        mt = mt < Mt ? mt : Mt - 1;            // clamp (duplicate tile writes same data)
        const int row = mt * 16 + l15;
        v8f acc = {};
        #pragma unroll
        for (int kc = 0; kc < KC; ++kc) {
            // A fragment: halves 0..15 = K kbaseA..+15
            const _Float16* ap = A + (size_t)row * K + (kc << 5) + kbaseA;
            v16h a = *(const v16h*)(ap);
            acc = __builtin_amdgcn_wmma_f32_16x16x32_f16(
                /*neg_a=*/false, a, /*neg_b=*/false, bf[kc],
                /*c_mod=*/(short)0, acc, /*reuse_a=*/false, /*reuse_b=*/false);
        }
        const int rbase = mt * 16 + ((lane < 16) ? 0 : 8);  // C/D: VGPR rr -> M=rbase+rr
        #pragma unroll
        for (int rr = 0; rr < 8; ++rr)
            C[(size_t)(rbase + rr) * Nout + nt * 16 + l15] = acc[rr] + bv;
    }
}

// ================= elementwise / encoder kernels =================
__global__ void lmp_fill_zero(float* __restrict__ p, int n) {
    int i = blockIdx.x * blockDim.x + threadIdx.x;
    if (i < n) p[i] = 0.0f;
}

__global__ void lmp_atom_encoder(const int* __restrict__ x,
                                 const float* __restrict__ atom_emb,
                                 float* __restrict__ h, int N) {
    int idx = blockIdx.x * blockDim.x + threadIdx.x;
    if (idx >= N * Hh) return;
    int n = idx >> 6, f = idx & 63;
    float s = 0.f;
    #pragma unroll
    for (int i = 0; i < 9; ++i) {
        int v = x[n * 9 + i];
        s += atom_emb[(size_t)(i * 100 + v) * Hh + f];
    }
    h[idx] = s;
}

__global__ void lmp_clique_gather_f16(const int* __restrict__ xc,
                                      const float* __restrict__ emb,
                                      _Float16* __restrict__ out, int NC) {
    int idx = blockIdx.x * blockDim.x + threadIdx.x;
    if (idx >= NC * Hh) return;
    int c = idx >> 6, f = idx & 63;
    out[idx] = (_Float16)emb[(size_t)xc[c] * Hh + f];
}

__global__ void lmp_cvt_f16(const float* __restrict__ in, _Float16* __restrict__ out, int n) {
    int i = blockIdx.x * blockDim.x + threadIdx.x;
    if (i < n) out[i] = (_Float16)in[i];
}

// GINE edge messages: agg[dst] += relu(h[src] + bond_emb sums). 16 threads/edge, 4 floats each.
__global__ void lmp_edge_msg(const int* __restrict__ ei, const int* __restrict__ ea,
                             const float* __restrict__ h, const float* __restrict__ bemb,
                             float* __restrict__ agg, int E) {
    long idx = (long)blockIdx.x * blockDim.x + threadIdx.x;
    if (idx >= (long)E * 16) return;
    int e = (int)(idx >> 4);
    int q = ((int)idx & 15) * 4;
    int src = ei[e];
    int dst = ei[E + e];
    int a0 = ea[e * 3 + 0], a1 = ea[e * 3 + 1], a2 = ea[e * 3 + 2];
    const float* hv = h + (size_t)src * Hh + q;
    const float* b0 = bemb + (size_t)(0 * 6 + a0) * Hh + q;
    const float* b1 = bemb + (size_t)(1 * 6 + a1) * Hh + q;
    const float* b2 = bemb + (size_t)(2 * 6 + a2) * Hh + q;
    float* out = agg + (size_t)dst * Hh + q;
    #pragma unroll
    for (int j = 0; j < 4; ++j) {
        float m = hv[j] + b0[j] + b1[j] + b2[j];
        m = m > 0.f ? m : 0.f;
        atomicAdd(out + j, m);
    }
}

// z = (1+eps)*h + agg, emitted as f16 GEMM operand
__global__ void lmp_combine_f16(const float* __restrict__ h, const float* __restrict__ agg,
                                const float* __restrict__ eps, int li,
                                _Float16* __restrict__ out, int n) {
    int i = blockIdx.x * blockDim.x + threadIdx.x;
    if (i >= n) return;
    float e = eps[li];
    out[i] = (_Float16)((1.0f + e) * h[i] + agg[i]);
}

// training-mode BatchNorm stats (biased variance): one block (256 thr) per channel
__global__ void lmp_bn_stats(const float* __restrict__ A, int M, int C,
                             float* __restrict__ stats) {
    __shared__ float ss[256];
    __shared__ float s2[256];
    int c = blockIdx.x;
    int tid = threadIdx.x;
    float s = 0.f, q = 0.f;
    for (int r = tid; r < M; r += 256) {
        float v = A[(size_t)r * C + c];
        s += v; q += v * v;
    }
    ss[tid] = s; s2[tid] = q;
    __syncthreads();
    for (int o = 128; o > 0; o >>= 1) {
        if (tid < o) { ss[tid] += ss[tid + o]; s2[tid] += s2[tid + o]; }
        __syncthreads();
    }
    if (tid == 0) {
        float mean = ss[0] / (float)M;
        float var  = s2[0] / (float)M - mean * mean;
        stats[c]       = mean;
        stats[128 + c] = rsqrtf(var + 1e-5f);
    }
}

__global__ void lmp_bn_apply_relu_f16(const float* __restrict__ A, const float* __restrict__ g,
                                      const float* __restrict__ b, const float* __restrict__ stats,
                                      _Float16* __restrict__ out, int M, int C) {
    int i = blockIdx.x * blockDim.x + threadIdx.x;
    if (i >= M * C) return;
    int c = i % C;
    float v = (A[i] - stats[c]) * stats[128 + c] * g[c] + b[c];
    out[i] = (_Float16)(v > 0.f ? v : 0.f);
}

__global__ void lmp_bn_apply_f32(const float* __restrict__ A, const float* __restrict__ g,
                                 const float* __restrict__ b, const float* __restrict__ stats,
                                 float* __restrict__ out, int M, int C, int do_relu) {
    int i = blockIdx.x * blockDim.x + threadIdx.x;
    if (i >= M * C) return;
    int c = i % C;
    float v = (A[i] - stats[c]) * stats[128 + c] * g[c] + b[c];
    if (do_relu) v = v > 0.f ? v : 0.f;
    out[i] = v;
}

__global__ void lmp_hc_update(float* __restrict__ hc, const float* __restrict__ t, int n) {
    int i = blockIdx.x * blockDim.x + threadIdx.x;
    if (i >= n) return;
    float v = t[i];
    hc[i] += (v > 0.f ? v : 0.f);
}

__global__ void lmp_h_update_lrelu(float* __restrict__ h, const float* __restrict__ t, int n) {
    int i = blockIdx.x * blockDim.x + threadIdx.x;
    if (i >= n) return;
    float v = t[i];
    h[i] += (v > 0.f ? v : 0.01f * v);
}

// generic 64-wide segment scatter-add: dst[didx[i]] += src[sidx[i] or i]
__global__ void lmp_scatter_add64(const float* __restrict__ srcbuf, const int* __restrict__ sidx,
                                  const int* __restrict__ didx, float* __restrict__ dst, int n) {
    long idx = (long)blockIdx.x * blockDim.x + threadIdx.x;
    if (idx >= (long)n * 16) return;
    int i = (int)(idx >> 4);
    int q = ((int)idx & 15) * 4;
    int si = sidx ? sidx[i] : i;
    const float* s = srcbuf + (size_t)si * Hh + q;
    float* d = dst + (size_t)didx[i] * Hh + q;
    #pragma unroll
    for (int j = 0; j < 4; ++j) atomicAdd(d + j, s[j]);
}

// ================= host-side orchestration =================
static inline int gUp(long n, int b) { return (int)((n + b - 1) / b); }
static inline dim3 gemmGrid(int M, int Nout) { return dim3(((M >> 4) + 1) / 2, Nout >> 4); }

extern "C" void kernel_launch(void* const* d_in, const int* in_sizes, int n_in,
                              void* d_out, int out_size, void* d_ws, size_t ws_size,
                              hipStream_t stream) {
    // inputs (setup_inputs order)
    const int*   x         = (const int*)d_in[0];
    const int*   x_clique  = (const int*)d_in[1];
    const int*   edge_attr = (const int*)d_in[2];
    const int*   edge_index= (const int*)d_in[3];
    const int*   a2c_row   = (const int*)d_in[4];
    const int*   a2c_col   = (const int*)d_in[5];
    const int*   batch     = (const int*)d_in[6];
    // d_in[7] graph_pe: unused by reference
    const float* atom_emb  = (const float*)d_in[8];
    const float* clique_emb= (const float*)d_in[9];
    const float* clique_W  = (const float*)d_in[10];
    const float* clique_b  = (const float*)d_in[11];
    const float* bond_emb  = (const float*)d_in[12];
    const float* eps       = (const float*)d_in[13];
    const float* W1        = (const float*)d_in[14];
    const float* b1        = (const float*)d_in[15];
    const float* bn1_g     = (const float*)d_in[16];
    const float* bn1_b     = (const float*)d_in[17];
    const float* W2        = (const float*)d_in[18];
    const float* b2        = (const float*)d_in[19];
    const float* gn_g      = (const float*)d_in[20];
    const float* gn_b      = (const float*)d_in[21];
    const float* sn_g      = (const float*)d_in[22];
    const float* sn_b      = (const float*)d_in[23];
    const float* a2c_W     = (const float*)d_in[24];
    const float* a2c_b     = (const float*)d_in[25];
    const float* c2a_W     = (const float*)d_in[26];
    const float* c2a_b     = (const float*)d_in[27];

    float* hc = (float*)d_out;                 // [NC,64]
    float* rd = hc + (size_t)NCc * Hh;         // [G,64]

    // workspace carve (256B aligned regions)
    char* w = (char*)d_ws;
    size_t o = 0;
    auto carve = [&](size_t bytes) { void* p = w + o; o += (bytes + 255) & ~(size_t)255; return p; };
    float*    h     = (float*)carve((size_t)Nn  * Hh  * 4);   // node features
    float*    y1    = (float*)carve((size_t)Nn  * 128 * 4);   // MLP mid / c2a result
    float*    agg   = (float*)carve((size_t)Nn  * Hh  * 4);   // edge agg / y2 / a_agg
    float*    c_agg = (float*)carve((size_t)NCc * Hh  * 4);
    float*    tmpc  = (float*)carve((size_t)NCc * Hh  * 4);
    _Float16* t16   = (_Float16*)carve((size_t)Nn * 128 * 2); // f16 GEMM A staging
    _Float16* w16   = (_Float16*)carve((size_t)128 * 64 * 2); // packed f16 weight staging
    float*    stats = (float*)carve(256 * 4);
    (void)ws_size; (void)in_sizes; (void)n_in; (void)out_size;

    const int B = 256;

    // ---- encoders ----
    lmp_atom_encoder<<<gUp((long)Nn * Hh, B), B, 0, stream>>>(x, atom_emb, h, Nn);
    lmp_clique_gather_f16<<<gUp((long)NCc * Hh, B), B, 0, stream>>>(x_clique, clique_emb, t16, NCc);
    lmp_pack_w16<<<gUp(64 * 64, B), B, 0, stream>>>(clique_W, w16, 64, 64);
    lmp_gemm_wmma<64><<<gemmGrid(NCc, 64), 32, 0, stream>>>(t16, w16, clique_b, hc, NCc, 64);

    for (int i = 0; i < Ll; ++i) {
        // GINEConv aggregate
        lmp_fill_zero<<<gUp((long)Nn * Hh, B), B, 0, stream>>>(agg, Nn * Hh);
        lmp_edge_msg<<<gUp((long)Ee * 16, B), B, 0, stream>>>(
            edge_index, edge_attr, h, bond_emb + (size_t)i * 3 * 6 * Hh, agg, Ee);
        lmp_combine_f16<<<gUp((long)Nn * Hh, B), B, 0, stream>>>(h, agg, eps, i, t16, Nn * Hh);
        // Linear(64->128) + BN + ReLU
        lmp_pack_w16<<<gUp(64 * 128, B), B, 0, stream>>>(W1 + (size_t)i * 64 * 128, w16, 64, 128);
        lmp_gemm_wmma<64><<<gemmGrid(Nn, 128), 32, 0, stream>>>(t16, w16, b1 + i * 128, y1, Nn, 128);
        lmp_bn_stats<<<128, 256, 0, stream>>>(y1, Nn, 128, stats);
        lmp_bn_apply_relu_f16<<<gUp((long)Nn * 128, B), B, 0, stream>>>(
            y1, bn1_g + i * 128, bn1_b + i * 128, stats, t16, Nn, 128);
        // Linear(128->64) + graph-norm + ReLU -> h
        lmp_pack_w16<<<gUp(128 * 64, B), B, 0, stream>>>(W2 + (size_t)i * 128 * 64, w16, 128, 64);
        lmp_gemm_wmma<128><<<gemmGrid(Nn, 64), 32, 0, stream>>>(t16, w16, b2 + i * 64, agg, Nn, 64);
        lmp_bn_stats<<<64, 256, 0, stream>>>(agg, Nn, 64, stats);
        lmp_bn_apply_f32<<<gUp((long)Nn * Hh, B), B, 0, stream>>>(
            agg, gn_g + i * 64, gn_b + i * 64, stats, h, Nn, 64, 1);
        // atom -> clique
        lmp_fill_zero<<<gUp((long)NCc * Hh, B), B, 0, stream>>>(c_agg, NCc * Hh);
        lmp_scatter_add64<<<gUp((long)Nn * 16, B), B, 0, stream>>>(h, a2c_row, a2c_col, c_agg, Nn);
        lmp_cvt_f16<<<gUp((long)NCc * Hh, B), B, 0, stream>>>(c_agg, t16, NCc * Hh);
        lmp_pack_w16<<<gUp(64 * 64, B), B, 0, stream>>>(a2c_W + (size_t)i * 64 * 64, w16, 64, 64);
        lmp_gemm_wmma<64><<<gemmGrid(NCc, 64), 32, 0, stream>>>(t16, w16, a2c_b + i * 64, tmpc, NCc, 64);
        lmp_hc_update<<<gUp((long)NCc * Hh, B), B, 0, stream>>>(hc, tmpc, NCc * Hh);
        lmp_bn_stats<<<64, 256, 0, stream>>>(hc, NCc, 64, stats);
        lmp_bn_apply_f32<<<gUp((long)NCc * Hh, B), B, 0, stream>>>(
            hc, sn_g + i * 64, sn_b + i * 64, stats, hc, NCc, 64, 0);
        // clique -> atom
        lmp_fill_zero<<<gUp((long)Nn * Hh, B), B, 0, stream>>>(agg, Nn * Hh);
        lmp_scatter_add64<<<gUp((long)Nn * 16, B), B, 0, stream>>>(hc, a2c_col, a2c_row, agg, Nn);
        lmp_cvt_f16<<<gUp((long)Nn * Hh, B), B, 0, stream>>>(agg, t16, Nn * Hh);
        lmp_pack_w16<<<gUp(64 * 64, B), B, 0, stream>>>(c2a_W + (size_t)i * 64 * 64, w16, 64, 64);
        lmp_gemm_wmma<64><<<gemmGrid(Nn, 64), 32, 0, stream>>>(t16, w16, c2a_b + i * 64, y1, Nn, 64);
        lmp_h_update_lrelu<<<gUp((long)Nn * Hh, B), B, 0, stream>>>(h, y1, Nn * Hh);
    }

    // readout
    lmp_fill_zero<<<gUp((long)Gg * Hh, B), B, 0, stream>>>(rd, Gg * Hh);
    lmp_scatter_add64<<<gUp((long)Nn * 16, B), B, 0, stream>>>(h, nullptr, batch, rd, Nn);
}
